// BatchKmeans_88819923681437
// MI455X (gfx1250) — matrix operand order repaired and checked by reading, not compiled, not run on hardware
//
#include <hip/hip_runtime.h>
#include <hip/hip_bf16.h>
#include <math.h>
#include <stdint.h>

// Problem constants (from the reference): N=131072 points, K=1024 centroids, D=256 dims.
#define N_PTS 131072
#define K_CB  1024
#define D_DIM 256

typedef __attribute__((ext_vector_type(16))) _Float16 v16h;
typedef __attribute__((ext_vector_type(8)))  float    v8f;

// Workspace layout (float units):
//   ws[0]                      : global distance-sum accumulator
//   ws[WS_XSQ  .. +N)          : x_sq per point
//   ws[WS_CSQ  .. +K)          : c_sq per centroid
//   ws[WS_BUCK .. +N)   (int)  : bucket (argmin) per point
//   ws[WS_CBH  .. +K*D/2) (f16): codebook hi halves, row-major [K][D]
//   ws[WS_CBL  .. +K*D/2) (f16): codebook lo halves, row-major [K][D]
#define WS_XSQ   16
#define WS_CSQ   (WS_XSQ + N_PTS)
#define WS_BUCK  (WS_CSQ + K_CB)
#define WS_CBH   (WS_BUCK + N_PTS)
#define WS_CBL   (WS_CBH + (K_CB * D_DIM / 2))

// LDS staging: 16 centroid rows x 256 halfs, padded to 272 halfs (32B pad)
// to avoid 512B-stride bank conflicts. Double buffered, hi + lo planes.
#define ROWP 272

// ---------------------------------------------------------------------------
// Kernel 1: exact fp32 squared norms, codebook f16 hi/lo split, output seed.
// Grid covers K*D threads (>= N).
// ---------------------------------------------------------------------------
__global__ void km_init(const float* __restrict__ x, const float* __restrict__ cb,
                        const float* __restrict__ bins, const float* __restrict__ nums,
                        float* __restrict__ out, float* __restrict__ ws) {
    const int i = blockIdx.x * blockDim.x + threadIdx.x;
    _Float16* cbh = (_Float16*)(ws + WS_CBH);
    _Float16* cbl = (_Float16*)(ws + WS_CBL);

    if (i < K_CB * D_DIM) {
        out[1 + i] = bins[i];                 // new_bins starts at bins
        const float v  = cb[i];               // hi/lo split: v ~= hi + lo (f16 pair)
        const _Float16 hi = (_Float16)v;
        cbh[i] = hi;
        cbl[i] = (_Float16)(v - (float)hi);
    }
    if (i < K_CB) {
        out[1 + K_CB * D_DIM + i] = nums[i];  // new_nums starts at nums
        const float4* row = (const float4*)(cb + (size_t)i * D_DIM);
        float s = 0.0f;
        #pragma unroll 8
        for (int d = 0; d < D_DIM / 4; ++d) {
            float4 v = row[d];
            s += v.x * v.x + v.y * v.y + v.z * v.z + v.w * v.w;
        }
        ws[WS_CSQ + i] = s;
    }
    if (i < N_PTS) {
        const float4* row = (const float4*)(x + (size_t)i * D_DIM);
        float s = 0.0f;
        #pragma unroll 8
        for (int d = 0; d < D_DIM / 4; ++d) {
            float4 v = row[d];
            s += v.x * v.x + v.y * v.y + v.z * v.z + v.w * v.w;
        }
        ws[WS_XSQ + i] = s;
    }
    if (i == 0) ws[0] = 0.0f;
}

// ---------------------------------------------------------------------------
// Kernel 2: WMMA distance + argmin. 4 waves/block; each wave owns 16 x-rows.
// B tiles (16 centroids, hi+lo f16) are staged global->LDS with the CDNA5
// async-LDS path (ASYNCcnt), double-buffered, and shared by all 4 waves.
// Split-precision f16 GEMM: dot = hi*hi + hi*lo + lo*hi (f32 accumulate).
// ---------------------------------------------------------------------------
__device__ __forceinline__ void async_tile_load(const _Float16* __restrict__ cbh,
                                                const _Float16* __restrict__ cbl,
                                                _Float16* ldsbuf, int colBase, int tid) {
    // Tile = 16 rows x 256 halfs, hi plane then lo plane. 1024 16B chunks.
    // 128 threads x 8 chunks; each source-line execution moves 512B per wave.
    #pragma unroll
    for (int it = 0; it < 8; ++it) {
        const int q    = it * 128 + tid;          // global chunk id
        const int plane = q >> 9;                 // 0 = hi, 1 = lo
        const int qq   = q & 511;
        const int row  = qq >> 5;                 // centroid row in tile
        const int off  = (qq & 31) * 8;           // half offset within row
        const _Float16* src = (plane ? cbl : cbh) + (size_t)(colBase + row) * D_DIM + off;
        _Float16* dst = ldsbuf + (size_t)plane * 16 * ROWP + row * ROWP + off;
        const unsigned lds_addr = (unsigned)(uintptr_t)dst;
        const unsigned long long gaddr = (unsigned long long)(uintptr_t)src;
        asm volatile("global_load_async_to_lds_b128 %0, %1, off"
                     :: "v"(lds_addr), "v"(gaddr) : "memory");
    }
}

__global__ void __launch_bounds__(128)
km_assign(const float* __restrict__ x, float* __restrict__ ws) {
    __shared__ _Float16 lds_tile[2][2 * 16 * ROWP];   // [buffer][plane*16*ROWP]

    const int tid     = threadIdx.x;
    const int lane    = tid & 31;
    const int wave    = tid >> 5;
    const int rowBase = (blockIdx.x * 4 + wave) * 16;

    const float* xsq = ws + WS_XSQ;
    const float* csq = ws + WS_CSQ;
    int* buckets     = (int*)(ws + WS_BUCK);
    const _Float16* cbh = (const _Float16*)(ws + WS_CBH);
    const _Float16* cbl = (const _Float16*)(ws + WS_CBL);

    const int halfSel = lane >> 4;      // 0: lanes 0-15, 1: lanes 16-31
    const int lrow    = lane & 15;
    const int aRow    = rowBase + lrow; // A layout: M = lane&15 in both halves
    const int kbaseA  = halfSel * 8;    // A layout: K base 0 / 8 per half-wave

    // Hoist the A operand (16 rows x 256 dims) into registers: 8 chunks of 16x32.
    // Converted once per wave, amortized over the 64-column-tile loop.
    v16h a_hi[8], a_lo[8];
    #pragma unroll
    for (int c = 0; c < 8; ++c) {
        #pragma unroll
        for (int v = 0; v < 8; ++v) {
            // A k order chosen here: K = halfSel*16 + 2*v + e (contiguous, matches B)
            const int kOff = halfSel * 16 + 2 * v;
            float2 f = *(const float2*)(x + (size_t)aRow * D_DIM + c * 32 + kOff);
            _Float16 hx = (_Float16)f.x, hy = (_Float16)f.y;
            a_hi[c][2 * v]     = hx;
            a_hi[c][2 * v + 1] = hy;
            a_lo[c][2 * v]     = (_Float16)(f.x - (float)hx);
            a_lo[c][2 * v + 1] = (_Float16)(f.y - (float)hy);
        }
    }

    // x_sq for the 8 rows this lane's C registers cover (M = r + halfSel*8).
    float xs[8];
    #pragma unroll
    for (int r = 0; r < 8; ++r) xs[r] = xsq[rowBase + r + halfSel * 8];

    float bestD2[8];
    int   bestIdx[8];
    #pragma unroll
    for (int r = 0; r < 8; ++r) { bestD2[r] = 3.4e38f; bestIdx[r] = 0; }
    float distSum = 0.0f;

    // Prologue: async-stage tile 0 into buffer 0.
    async_tile_load(cbh, cbl, &lds_tile[0][0], 0, tid);

    for (int kt = 0; kt < K_CB / 16; ++kt) {
        const int cur     = kt & 1;
        const int colBase = kt * 16;
        const int col     = colBase + lrow;      // B/C layout: N = lane&15

        // Keep the pipeline full: stage tile kt+1 into the other buffer.
        // (End-of-iteration barrier of kt-1 made that buffer safe to overwrite.)
        if (kt + 1 < K_CB / 16)
            async_tile_load(cbh, cbl, &lds_tile[cur ^ 1][0], colBase + 16, tid);

        // Tile kt's 8 async loads are the oldest outstanding; async loads
        // complete in order, so <=8 pending means tile kt is fully in LDS.
        asm volatile("s_wait_asynccnt 0x8" ::: "memory");
        __syncthreads();

        const float cs = csq[col];
        // B layout: lane reads 16 contiguous halfs of centroid row `lrow`
        // at K = c*32 + halfSel*16  ->  32B-aligned LDS vector loads.
        const _Float16* bh = &lds_tile[cur][(size_t)lrow * ROWP + halfSel * 16];
        const _Float16* bl = bh + 16 * ROWP;

        v8f C = {};
        #pragma unroll
        for (int c = 0; c < 8; ++c) {
            v16h b_hi = *(const v16h*)(bh + c * 32);
            v16h b_lo = *(const v16h*)(bl + c * 32);
            // dot ~= hi*hi + hi*lo + lo*hi  (near-fp32 accuracy, f32 accumulate)
            C = __builtin_amdgcn_wmma_f32_16x16x32_f16(false, a_hi[c], false, b_hi,
                                                       (short)0, C, false, false);
            C = __builtin_amdgcn_wmma_f32_16x16x32_f16(false, a_hi[c], false, b_lo,
                                                       (short)0, C, false, false);
            C = __builtin_amdgcn_wmma_f32_16x16x32_f16(false, a_lo[c], false, b_hi,
                                                       (short)0, C, false, false);
        }

        // Epilogue on the 16x16 tile: d2 = |x|^2 + |c|^2 - 2 x.c
        #pragma unroll
        for (int r = 0; r < 8; ++r) {
            float d2 = fmaxf(xs[r] + cs - 2.0f * C[r], 0.0f);
            distSum += sqrtf(d2);                // sqrt runs on the TRANS pipe
            if (d2 < bestD2[r]) { bestD2[r] = d2; bestIdx[r] = col; }
        }

        // All waves done reading buffer `cur` before it is refilled at kt+2.
        __syncthreads();
    }

    // Argmin reduce across the 16 lanes (columns) covering the same rows.
    // Tie-break on the smaller index to match jnp.argmin (first occurrence).
    #pragma unroll
    for (int off = 8; off >= 1; off >>= 1) {
        #pragma unroll
        for (int r = 0; r < 8; ++r) {
            float od2 = __shfl_xor(bestD2[r], off);
            int   oid = __shfl_xor(bestIdx[r], off);
            if (od2 < bestD2[r] || (od2 == bestD2[r] && oid < bestIdx[r])) {
                bestD2[r] = od2;
                bestIdx[r] = oid;
            }
        }
    }
    if (lrow == 0) {
        #pragma unroll
        for (int r = 0; r < 8; ++r)
            buckets[rowBase + r + halfSel * 8] = bestIdx[r];
    }

    // Wave-wide distance sum -> single global atomic per wave.
    #pragma unroll
    for (int off = 16; off >= 1; off >>= 1) distSum += __shfl_xor(distSum, off);
    if (lane == 0) atomicAdd(ws, distSum);
}

// ---------------------------------------------------------------------------
// Kernel 3: segment-sum scatter. One wave per point row.
// ---------------------------------------------------------------------------
__global__ void km_scatter(const float* __restrict__ x, const float* __restrict__ ws,
                           float* __restrict__ out) {
    const int* buckets = (const int*)(ws + WS_BUCK);
    const int lane = threadIdx.x & 31;
    const int wave = threadIdx.x >> 5;
    const int row  = blockIdx.x * (blockDim.x >> 5) + wave;
    if (row >= N_PTS) return;

    const int b = buckets[row];
    float* bins = out + 1;
    float* nums = out + 1 + K_CB * D_DIM;
    const float* xr = x + (size_t)row * D_DIM;
    #pragma unroll
    for (int d = lane; d < D_DIM; d += 32)
        atomicAdd(&bins[(size_t)b * D_DIM + d], xr[d]);
    if (lane == 0) atomicAdd(&nums[b], 1.0f);
}

// ---------------------------------------------------------------------------
// Kernel 4: mean distance.
// ---------------------------------------------------------------------------
__global__ void km_final(const float* __restrict__ ws, float* __restrict__ out) {
    if (threadIdx.x == 0 && blockIdx.x == 0)
        out[0] = ws[0] / (float)((long long)N_PTS * (long long)K_CB);
}

extern "C" void kernel_launch(void* const* d_in, const int* in_sizes, int n_in,
                              void* d_out, int out_size, void* d_ws, size_t ws_size,
                              hipStream_t stream) {
    const float* x    = (const float*)d_in[0];
    const float* cb   = (const float*)d_in[1];
    const float* bins = (const float*)d_in[2];
    const float* nums = (const float*)d_in[3];
    float* out = (float*)d_out;
    float* ws  = (float*)d_ws;

    // 1) norms + codebook hi/lo split + output seed (grid covers K*D >= N)
    km_init<<<(K_CB * D_DIM + 255) / 256, 256, 0, stream>>>(x, cb, bins, nums, out, ws);
    // 2) WMMA distances + argmin: 4 waves/block, 16 rows/wave, async-LDS B tiles
    km_assign<<<N_PTS / 64, 128, 0, stream>>>(x, ws);
    // 3) scatter-add into bins/nums: 8 waves/block, 1 row/wave
    km_scatter<<<N_PTS / 8, 256, 0, stream>>>(x, ws, out);
    // 4) mean distance
    km_final<<<1, 1, 0, stream>>>(ws, out);
}
